// GLACrossAttention_23124103921714
// MI455X (gfx1250) — matrix operand order repaired and checked by reading, not compiled, not run on hardware
//
#include <hip/hip_runtime.h>

// ---------------------------------------------------------------------------
// GLA cross-attention for MI455X (gfx1250, wave32, WMMA 16x16x32 bf16,
// TDM tensor_load_to_lds + double-buffered pipeline + global_prefetch)
// ---------------------------------------------------------------------------

typedef __attribute__((ext_vector_type(16))) __bf16 v16bf;
typedef __attribute__((ext_vector_type(8)))  __bf16 v8bf;
typedef __attribute__((ext_vector_type(8)))  float  v8f;
typedef __attribute__((ext_vector_type(4)))  unsigned int u32x4;
typedef __attribute__((ext_vector_type(8)))  int i32x8;
typedef __attribute__((ext_vector_type(4)))  int i32x4;

#define DEVINL __device__ __forceinline__

constexpr int B_   = 2;
constexpr int QL_  = 1024;
constexpr int KL_  = 4096;
constexpr int HID_ = 2048;
constexpr int H_   = 16;
constexpr int HD_  = 128;
constexpr int RD_  = 64;
constexpr int G_   = 4;
constexpr int QLR_ = 512;
constexpr int LAT_ = 256;
constexpr int HRD_  = HD_ + RD_;      // 192
constexpr int UQN_  = H_ * HRD_;      // 3072
constexpr int DKVN_ = G_*LAT_ + RD_;  // 1088
constexpr int KVN_  = G_*LAT_;        // 1024
constexpr int KVW_  = H_ * 2 * HD_;   // 4096 (per-token kv row)

#if __has_builtin(__builtin_amdgcn_tensor_load_to_lds) && \
    __has_builtin(__builtin_amdgcn_s_wait_tensorcnt)
#define HAVE_TDM 1
#else
#define HAVE_TDM 0
#endif

// 16-bit A-matrix (16x32) K index for half-element j (0..15) in lane-half `half` (contiguous: k = half*16 + j)
DEVINL int a16_k(int j, int half) {
  return half*16 + j;
}

DEVINL v8f wmma_bf16(v16bf a, v16bf b, v8f c) {
  return __builtin_amdgcn_wmma_f32_16x16x32_bf16(false, a, false, b, (short)0, c,
                                                 false, false);
}

// Build a 16x32 A-fragment from two contiguous 8-half LDS chunks.
DEVINL v16bf a_frag(const __bf16* base, int half) {
  v8bf lo = *(const v8bf*)(base + half * 16);
  v8bf hi = *(const v8bf*)(base + half * 16 + 8);
  return __builtin_shufflevector(lo, hi, 0,1,2,3,4,5,6,7,8,9,10,11,12,13,14,15);
}

// ---------------------------------------------------------------------------
// fp32 -> bf16 conversion (grid-stride)
// ---------------------------------------------------------------------------
__global__ void k_cvt_bf16(const float* __restrict__ src, __bf16* __restrict__ dst,
                           long long n) {
  long long i = (long long)blockIdx.x * blockDim.x + threadIdx.x;
  long long stride = (long long)gridDim.x * blockDim.x;
  for (; i < n; i += stride) dst[i] = (__bf16)src[i];
}

// fp32 (K x N) -> bf16 transposed (N x K), batched; 32x32 LDS tile transpose.
__global__ __launch_bounds__(256) void k_cvt_t_bf16(const float* __restrict__ src,
                                                    __bf16* __restrict__ dst,
                                                    int K, int N) {
  __shared__ float t[32][33];
  const int tx = threadIdx.x & 31, ty = threadIdx.x >> 5;
  const float* s = src + (size_t)blockIdx.z * K * N;
  __bf16* d = dst + (size_t)blockIdx.z * K * N;
  const int k0 = blockIdx.y * 32, n0 = blockIdx.x * 32;
#pragma unroll
  for (int i = 0; i < 32; i += 8)
    t[ty + i][tx] = s[(size_t)(k0 + ty + i) * N + n0 + tx];
  __syncthreads();
#pragma unroll
  for (int i = 0; i < 32; i += 8)
    d[(size_t)(n0 + ty + i) * K + k0 + tx] = (__bf16)t[tx][ty + i];
}

// ---------------------------------------------------------------------------
// RMSNorms
// ---------------------------------------------------------------------------
__global__ __launch_bounds__(256) void k_rmsnorm_q(const float* __restrict__ x,
                                                   const float* __restrict__ w,
                                                   __bf16* __restrict__ out) {
  __shared__ float red[256];
  const int row = blockIdx.x, tid = threadIdx.x;
  const float* xr = x + (size_t)row * QLR_;
  float ss = 0.f;
  for (int i = tid; i < QLR_; i += 256) { float v = xr[i]; ss += v * v; }
  red[tid] = ss; __syncthreads();
  for (int s = 128; s > 0; s >>= 1) {
    if (tid < s) red[tid] += red[tid + s];
    __syncthreads();
  }
  const float scale = rsqrtf(red[0] / (float)QLR_ + 1e-6f);
  __bf16* orow = out + (size_t)row * QLR_;
  for (int i = tid; i < QLR_; i += 256) orow[i] = (__bf16)(xr[i] * scale * w[i]);
}

__global__ __launch_bounds__(256) void k_rmsnorm_kv(const float* __restrict__ ckv,
                                                    const float* __restrict__ w,
                                                    __bf16* __restrict__ out) {
  __shared__ float red[256];
  const int rowg = blockIdx.x;
  const int t = rowg >> 2, g = rowg & 3;
  const int tid = threadIdx.x;
  const float v = ckv[(size_t)t * DKVN_ + g * LAT_ + tid];
  red[tid] = v * v; __syncthreads();
  for (int s = 128; s > 0; s >>= 1) {
    if (tid < s) red[tid] += red[tid + s];
    __syncthreads();
  }
  const float scale = rsqrtf(red[0] / (float)LAT_ + 1e-6f);
  out[(size_t)t * KVN_ + g * LAT_ + tid] = (__bf16)(v * scale * w[g * LAT_ + tid]);
}

// ---------------------------------------------------------------------------
// Software-pipelined LDS-staged WMMA GEMM:  C(M,N) = A(M,K) * Bt(N,K)^T
// Block tile 128x128, 8 waves in 2(M) x 4(N), 4x2 WMMA tiles per wave.
// Next K-stage is fetched into registers while the current stage computes;
// stage+2 lines are touched with global_prefetch_b8.
// ---------------------------------------------------------------------------
constexpr int GK  = 64;
constexpr int SAS = 72;   // 144B rows, 16B aligned
constexpr int SBS = 80;   // 160B rows, 32B aligned

template <typename OutT>
__global__ __launch_bounds__(256) void k_gemm_tn(const __bf16* __restrict__ A,
                                                 const __bf16* __restrict__ Bt,
                                                 OutT* __restrict__ C,
                                                 int M, int N, int K,
                                                 int lda, int ldb, int ldc) {
  __shared__ __attribute__((aligned(32))) __bf16 sA [128 * SAS];
  __shared__ __attribute__((aligned(32))) __bf16 sBt[128 * SBS];

  const int tid  = threadIdx.x;
  const int lane = tid & 31;
  const int wave = tid >> 5;
  const int half = lane >> 4;
  const int r    = lane & 15;
  const int wm   = wave >> 2;
  const int wn   = wave & 3;
  const int blockM = blockIdx.y * 128;
  const int blockN = blockIdx.x * 128;

  const int srow = tid >> 1;
  const int scol = (tid & 1) * 32;
  const int arow = blockM + srow;
  int brow = blockN + srow; if (brow >= N) brow = N - 1;
  const __bf16* gA = A  + (size_t)arow * lda + scol;
  const __bf16* gB = Bt + (size_t)brow * ldb + scol;

  v8f acc[4][2] = {};
  v8bf ra[4], rb[4];

#pragma unroll
  for (int i = 0; i < 4; ++i) {             // prologue: stage 0 -> regs
    ra[i] = *(const v8bf*)(gA + i * 8);
    rb[i] = *(const v8bf*)(gB + i * 8);
  }

  for (int k0 = 0; k0 < K; k0 += GK) {
#pragma unroll
    for (int i = 0; i < 4; ++i) {           // regs -> LDS
      *(v8bf*)&sA [srow * SAS + scol + i * 8] = ra[i];
      *(v8bf*)&sBt[srow * SBS + scol + i * 8] = rb[i];
    }
    __syncthreads();

    if (k0 + GK < K) {                      // fetch next stage while computing
#pragma unroll
      for (int i = 0; i < 4; ++i) {
        ra[i] = *(const v8bf*)(gA + (k0 + GK) + i * 8);
        rb[i] = *(const v8bf*)(gB + (k0 + GK) + i * 8);
      }
      if (k0 + 2 * GK < K) {
        __builtin_prefetch(gA + (k0 + 2 * GK), 0, 1);
        __builtin_prefetch(gB + (k0 + 2 * GK), 0, 1);
      }
    }

#pragma unroll
    for (int kk = 0; kk < GK; kk += 32) {
      v16bf af[4], bf[2];
#pragma unroll
      for (int mt = 0; mt < 4; ++mt)
        af[mt] = a_frag(&sA[(wm * 64 + mt * 16 + r) * SAS + kk], half);
#pragma unroll
      for (int nt = 0; nt < 2; ++nt)
        bf[nt] = *(const v16bf*)&sBt[(wn * 32 + nt * 16 + r) * SBS + kk + half * 16];
#pragma unroll
      for (int mt = 0; mt < 4; ++mt)
#pragma unroll
        for (int nt = 0; nt < 2; ++nt)
          acc[mt][nt] = wmma_bf16(af[mt], bf[nt], acc[mt][nt]);
    }
    __syncthreads();
  }

#pragma unroll
  for (int mt = 0; mt < 4; ++mt)
#pragma unroll
    for (int nt = 0; nt < 2; ++nt) {
      const int col = blockN + wn * 32 + nt * 16 + r;
      if (col < N) {
#pragma unroll
        for (int i = 0; i < 8; ++i)
          C[(size_t)(blockM + wm * 64 + mt * 16 + half * 8 + i) * ldc + col] =
              (OutT)acc[mt][nt][i];
      }
    }
}

// ---------------------------------------------------------------------------
// Flash attention, double-buffered 32-token chunks:
//   - K chunk DMA'd by the Tensor Data Mover (pad engine emits the 144-half
//     padded LDS rows the fragment reads expect); chunk c+1 issued while
//     chunk c computes, hidden behind s_wait_tensorcnt at the next barrier.
//   - V chunk transposed into LDS cooperatively (also pipelined one ahead).
// NOTE: all LDS buffer pointers are computed at point of use from smem_raw --
// storing them in arrays creates addrspacecast static initializers ld.lld
// cannot link.
// ---------------------------------------------------------------------------
constexpr int SKS = 144;
constexpr int STV = 48;
constexpr int SK_BYTES  = 32 * SKS * 2;    // 9216  (per buffer)
constexpr int SVT_BYTES = HD_ * STV * 2;   // 12288 (per buffer)
constexpr int SP_BYTES  = 8 * 16 * 32 * 2; // 8192
constexpr int VT_BASE   = 2 * SK_BYTES;            // 18432
constexpr int SP_BASE   = 2 * SK_BYTES + 2 * SVT_BYTES;  // 43008
constexpr int ATTN_LDS  = SP_BASE + SP_BYTES;      // 51200

DEVINL void tdm_issue_k(const __bf16* gk, unsigned ldsoff, i32x8 g1) {
#if HAVE_TDM
  const unsigned long long ga = (unsigned long long)(uintptr_t)gk;
  u32x4 g0;
  g0[0] = 1u;                                   // count=1 (valid user D#)
  g0[1] = ldsoff;                               // destination LDS byte offset
  g0[2] = (unsigned)(ga & 0xFFFFFFFFu);         // global_addr[31:0]
  g0[3] = (unsigned)((ga >> 32) & 0x1FFFFFFu) | (2u << 30);  // type=2
  const i32x4 gz = {};
#if __clang_major__ >= 23
  const i32x8 gz8 = {};
  __builtin_amdgcn_tensor_load_to_lds(g0, g1, gz, gz, gz8, 0);
#else
  __builtin_amdgcn_tensor_load_to_lds(g0, g1, gz, gz, 0);
#endif
#else
  (void)gk; (void)ldsoff; (void)g1;
#endif
}

__global__ __launch_bounds__(256) void k_attn(const __bf16* __restrict__ qfull,
                                              const __bf16* __restrict__ kv,
                                              __bf16* __restrict__ attn_out) {
  extern __shared__ __attribute__((aligned(64))) char smem_raw[];
  __bf16 (*sP)[16][32] = (__bf16(*)[16][32])(smem_raw + SP_BASE);

  const int tid  = threadIdx.x;
  const int lane = tid & 31;
  const int wave = tid >> 5;
  const int half = lane >> 4;
  const int r    = lane & 15;
  const int b  = blockIdx.x >> 4;
  const int h  = blockIdx.x & 15;
  const int qrow0 = blockIdx.y * 128 + wave * 16;
  const float SCALE = 0.08838834764831845f;  // 1/sqrt(128)
  const int NC = KL_ / 32;

  // Q A-fragments, loaded once.
  v16bf qf[4];
  const __bf16* qbase = qfull + (size_t)(b * QL_ + qrow0 + r) * UQN_ + h * HRD_;
#pragma unroll
  for (int s = 0; s < 4; ++s)
#pragma unroll
    for (int j = 0; j < 16; ++j) qf[s][j] = qbase[s * 32 + a16_k(j, half)];

  const __bf16* kbase = kv + (size_t)b * KL_ * KVW_ + h * 2 * HD_;
  const __bf16* vbase = kbase + HD_;

  v8f o[8] = {};
  float m[8], l[8];
#pragma unroll
  for (int i = 0; i < 8; ++i) { m[i] = -1e30f; l[i] = 0.f; }

#if HAVE_TDM
  // TDM descriptor group 1: data_size=2B; pad 8 DWORDs after every 64 DWORDs
  // (256B K row -> 144-half LDS rows); tensor/tile 128 x 32; dim0 stride 4096.
  i32x8 g1;
  g1[0] = (1 << 16) | (1 << 20) | (5 << 22) | (7 << 25);
  g1[1] = (HD_ & 0xFFFF) << 16;
  g1[2] = 32 << 16;
  g1[3] = (HD_ & 0xFFFF) << 16;
  g1[4] = 32;
  g1[5] = KVW_;
  g1[6] = 0;
  g1[7] = 0;
#endif

  const int srow = (tid * 16) >> 7;
  const int scol = (tid * 16) & 127;

  // ---- prologue: stage chunk 0 into buffer 0
#if HAVE_TDM
  if (wave == 0) tdm_issue_k(kbase, 0u, g1);
#else
  {
    __bf16* dK = (__bf16*)smem_raw;
    const __bf16* gk = kbase + (size_t)srow * KVW_ + scol;
#pragma unroll
    for (int i = 0; i < 16; ++i) dK[srow * SKS + scol + i] = gk[i];
  }
#endif
  {
    __bf16* dV = (__bf16*)(smem_raw + VT_BASE);
    const __bf16* gv = vbase + (size_t)srow * KVW_ + scol;
#pragma unroll
    for (int i = 0; i < 16; ++i) dV[(scol + i) * STV + srow] = gv[i];
  }

  for (int c = 0; c < NC; ++c) {
#if HAVE_TDM
    if (wave == 0) __builtin_amdgcn_s_wait_tensorcnt(0);
#endif
    __syncthreads();   // buffer c%2 ready for everyone; buffer (c+1)%2 free

    if (c + 1 < NC) {  // stage chunk c+1 into the other buffer (overlapped)
      const int n0 = (c + 1) * 32;
      const int nb = (c + 1) & 1;
#if HAVE_TDM
      if (wave == 0)
        tdm_issue_k(kbase + (size_t)n0 * KVW_, (unsigned)(nb * SK_BYTES), g1);
#else
      {
        __bf16* dK = (__bf16*)(smem_raw + nb * SK_BYTES);
        const __bf16* gk = kbase + (size_t)(n0 + srow) * KVW_ + scol;
#pragma unroll
        for (int i = 0; i < 16; ++i) dK[srow * SKS + scol + i] = gk[i];
      }
#endif
      {
        __bf16* dV = (__bf16*)(smem_raw + VT_BASE + nb * SVT_BYTES);
        const __bf16* gv = vbase + (size_t)(n0 + srow) * KVW_ + scol;
#pragma unroll
        for (int i = 0; i < 16; ++i) dV[(scol + i) * STV + srow] = gv[i];
      }
      if (c + 2 < NC)
        __builtin_prefetch(vbase + (size_t)((c + 2) * 32 + srow) * KVW_ + scol,
                           0, 1);
    }

    const __bf16* cK = (const __bf16*)(smem_raw + (c & 1) * SK_BYTES);
    const __bf16* cV = (const __bf16*)(smem_raw + VT_BASE + (c & 1) * SVT_BYTES);

    // S = Q K^T : batch all 8 B-fragments, then 2 independent 4-deep chains.
    v16bf bk[8];
#pragma unroll
    for (int t = 0; t < 2; ++t)
#pragma unroll
      for (int s = 0; s < 4; ++s)
        bk[t * 4 + s] =
            *(const v16bf*)&cK[(t * 16 + r) * SKS + s * 32 + half * 16];
    v8f sacc[2] = {};
#pragma unroll
    for (int s = 0; s < 4; ++s) {
      sacc[0] = wmma_bf16(qf[s], bk[s], sacc[0]);
      sacc[1] = wmma_bf16(qf[s], bk[4 + s], sacc[1]);
    }

    // online softmax; VGPR i in this lane is row (half*8 + i)
#pragma unroll
    for (int i = 0; i < 8; ++i) {
      float v0 = sacc[0][i] * SCALE, v1 = sacc[1][i] * SCALE;
      float mx = fmaxf(v0, v1);
#pragma unroll
      for (int d = 1; d < 16; d <<= 1) mx = fmaxf(mx, __shfl_xor(mx, d, 32));
      const float mn = fmaxf(m[i], mx);
      const float alpha = __expf(m[i] - mn);
      const float p0 = __expf(v0 - mn), p1 = __expf(v1 - mn);
      float rs = p0 + p1;
#pragma unroll
      for (int d = 1; d < 16; d <<= 1) rs += __shfl_xor(rs, d, 32);
      l[i] = l[i] * alpha + rs;
      m[i] = mn;
#pragma unroll
      for (int nt = 0; nt < 8; ++nt) o[nt][i] *= alpha;
      sP[wave][half * 8 + i][r]      = (__bf16)p0;
      sP[wave][half * 8 + i][16 + r] = (__bf16)p1;
    }

    // P as A-fragment (wave-private sP; LDS in-order within a wave)
    v16bf pf;
#pragma unroll
    for (int j = 0; j < 16; ++j) pf[j] = sP[wave][r][a16_k(j, half)];

    // O += P V : batch 8 V-fragments, then 8 independent chains.
    v16bf bv[8];
#pragma unroll
    for (int nt = 0; nt < 8; ++nt)
      bv[nt] = *(const v16bf*)&cV[(nt * 16 + r) * STV + half * 16];
#pragma unroll
    for (int nt = 0; nt < 8; ++nt) o[nt] = wmma_bf16(pf, bv[nt], o[nt]);
  }

#pragma unroll
  for (int i = 0; i < 8; ++i) {
    const float inv = 1.0f / l[i];
    const int orow = qrow0 + half * 8 + i;
#pragma unroll
    for (int nt = 0; nt < 8; ++nt)
      attn_out[(size_t)(b * QL_ + orow) * (H_ * HD_) + h * HD_ + nt * 16 + r] =
          (__bf16)(o[nt][i] * inv);
  }
}

// ---------------------------------------------------------------------------
// Host-side orchestration
// ---------------------------------------------------------------------------
static inline size_t align256(size_t x) { return (x + 255) & ~(size_t)255; }

extern "C" void kernel_launch(void* const* d_in, const int* in_sizes, int n_in,
                              void* d_out, int out_size, void* d_ws, size_t ws_size,
                              hipStream_t stream) {
  (void)in_sizes; (void)n_in; (void)out_size; (void)ws_size;
  const float* x_q    = (const float*)d_in[0];
  const float* x_kv   = (const float*)d_in[1];
  const float* W_dQ   = (const float*)d_in[2];
  const float* qnw    = (const float*)d_in[3];
  const float* W_uQ   = (const float*)d_in[4];
  const float* W_dKV  = (const float*)d_in[5];
  const float* kvnw   = (const float*)d_in[6];
  const float* W_ukv  = (const float*)d_in[7];
  const float* W_o    = (const float*)d_in[8];
  float* out = (float*)d_out;

  char* w = (char*)d_ws;
  size_t off = 0;
  auto take = [&](size_t elems, size_t esz) {
    void* p = w + off;
    off = align256(off + elems * esz);
    return p;
  };
  __bf16* xq_bf   = (__bf16*)take((size_t)B_*QL_*HID_, 2);
  __bf16* xkv_bf  = (__bf16*)take((size_t)B_*KL_*HID_, 2);
  __bf16* wdqT    = (__bf16*)take((size_t)HID_*QLR_, 2);
  __bf16* wuqT    = (__bf16*)take((size_t)QLR_*UQN_, 2);
  __bf16* wdkvT   = (__bf16*)take((size_t)HID_*DKVN_, 2);
  __bf16* wukvT   = (__bf16*)take((size_t)G_*LAT_*KVN_, 2);
  __bf16* woT     = (__bf16*)take((size_t)(H_*HD_)*HID_, 2);
  float*  qlat_f  = (float*) take((size_t)B_*QL_*QLR_, 4);
  __bf16* qlat_bf = (__bf16*)take((size_t)B_*QL_*QLR_, 2);
  __bf16* qfull_bf= (__bf16*)take((size_t)B_*QL_*UQN_, 2);
  float*  ckv_f   = (float*) take((size_t)B_*KL_*DKVN_, 4);
  __bf16* ckv_bf  = (__bf16*)take((size_t)B_*KL_*KVN_, 2);
  __bf16* kv_bf   = (__bf16*)take((size_t)B_*KL_*KVW_, 2);
  __bf16* attn_bf = (__bf16*)take((size_t)B_*QL_*(H_*HD_), 2);

  // 1) conversions: activations plain, weights transposed for the TN GEMM
  k_cvt_bf16<<<2048, 256, 0, stream>>>(x_q,  xq_bf,  (long long)B_*QL_*HID_);
  k_cvt_bf16<<<2048, 256, 0, stream>>>(x_kv, xkv_bf, (long long)B_*KL_*HID_);
  k_cvt_t_bf16<<<dim3(QLR_/32, HID_/32, 1), 256, 0, stream>>>(W_dQ,  wdqT,  HID_, QLR_);
  k_cvt_t_bf16<<<dim3(UQN_/32, QLR_/32, 1), 256, 0, stream>>>(W_uQ,  wuqT,  QLR_, UQN_);
  k_cvt_t_bf16<<<dim3(DKVN_/32, HID_/32, 1), 256, 0, stream>>>(W_dKV, wdkvT, HID_, DKVN_);
  k_cvt_t_bf16<<<dim3(KVN_/32, LAT_/32, G_), 256, 0, stream>>>(W_ukv, wukvT, LAT_, KVN_);
  k_cvt_t_bf16<<<dim3(HID_/32, (H_*HD_)/32, 1), 256, 0, stream>>>(W_o, woT, H_*HD_, HID_);

  const int MQ = B_ * QL_;   // 2048
  const int MK = B_ * KL_;   // 8192

  // 2) q_lat = x_q @ W_dQ  (2048 x 2048 x 512)
  k_gemm_tn<float><<<dim3(QLR_/128, MQ/128), 256, 0, stream>>>(
      xq_bf, wdqT, qlat_f, MQ, QLR_, HID_, HID_, HID_, QLR_);

  // 3) RMSNorm q -> bf16
  k_rmsnorm_q<<<MQ, 256, 0, stream>>>(qlat_f, qnw, qlat_bf);

  // 4) q_full = q_lat @ W_uQ  (2048 x 512 x 3072), bf16 out
  k_gemm_tn<__bf16><<<dim3(UQN_/128, MQ/128), 256, 0, stream>>>(
      qlat_bf, wuqT, qfull_bf, MQ, UQN_, QLR_, QLR_, QLR_, UQN_);

  // 5) ckv = x_kv @ W_dKV  (8192 x 2048 x 1088)
  k_gemm_tn<float><<<dim3((DKVN_ + 127)/128, MK/128), 256, 0, stream>>>(
      xkv_bf, wdkvT, ckv_f, MK, DKVN_, HID_, HID_, HID_, DKVN_);

  // 6) per-group RMSNorm (slices first 1024 cols) -> bf16
  k_rmsnorm_kv<<<MK * G_, 256, 0, stream>>>(ckv_f, kvnw, ckv_bf);

  // 7) kv = einsum('bsgl,gld') as 4 strided GEMMs (8192 x 256 x 1024), bf16
  for (int g = 0; g < G_; ++g) {
    k_gemm_tn<__bf16><<<dim3(KVN_/128, MK/128), 256, 0, stream>>>(
        ckv_bf + g * LAT_, wukvT + (size_t)g * KVN_ * LAT_, kv_bf + g * KVN_,
        MK, KVN_, LAT_, KVN_, LAT_, KVW_);
  }

  // 8) flash attention -> attn_bf (B*QL, 2048)
  k_attn<<<dim3(B_ * H_, QL_ / 128), 256, ATTN_LDS, stream>>>(qfull_bf, kv_bf,
                                                              attn_bf);

  // 9) out = attn @ W_o  (2048 x 2048 x 2048), f32 out
  k_gemm_tn<float><<<dim3(HID_/128, MQ/128), 256, 0, stream>>>(
      attn_bf, woT, out, MQ, HID_, H_*HD_, H_*HD_, H_*HD_, HID_);
}
